// ResourceAllocatorDecoder_42537356099733
// MI455X (gfx1250) — matrix operand order repaired and checked by reading, not compiled, not run on hardware
//
#include <hip/hip_runtime.h>
#include <hip/hip_bf16.h>
#include <cstdint>

#define BB 4
#define UU 1024
#define SS 256
#define FF 16
#define DD 256
#define KREDC 3
#define MAXSTEPS 32
#define NEGV (-1000000000.0f)

typedef __attribute__((ext_vector_type(16))) _Float16 v16h;
typedef __attribute__((ext_vector_type(8)))  float    v8f;

// ---------------- init: needs, cap, inv_init, fd_id ----------------
__global__ void k_init(const float* __restrict__ users, const float* __restrict__ servers,
                       const float* __restrict__ fd_onehot,
                       float* __restrict__ needs, float* __restrict__ cap,
                       float* __restrict__ inv_init, int* __restrict__ fd_id) {
  int gid = blockIdx.x * blockDim.x + threadIdx.x;
  if (gid < BB * UU) {
    #pragma unroll
    for (int c = 0; c < 4; c++) needs[gid * 4 + c] = users[gid * 8 + 2 + c];
  }
  if (gid < BB * SS) {
    #pragma unroll
    for (int c = 0; c < 4; c++) {
      float ic = servers[gid * 8 + 3 + c];
      cap[gid * 4 + c] = ic;
      inv_init[gid * 4 + c] = 1.0f / fmaxf(ic, 1e-6f);
    }
    int id = 0;
    for (int f = 0; f < FF; f++)
      if (fd_onehot[gid * FF + f] > 0.5f) id = f;
    fd_id[gid] = id;
  }
}

// ---------------- q = user_enc @ W_q^T  (block per (b,u), thread j) ----------------
__global__ void k_gemm_q(const float* __restrict__ ue, const float* __restrict__ Wq,
                         float* __restrict__ q) {
  __shared__ float row[DD];
  int bu = blockIdx.x;
  int j = threadIdx.x;
  row[j] = ue[bu * DD + j];
  __syncthreads();
  float acc = 0.0f;
  const float* w = Wq + j * DD;
  for (int d = 0; d < DD; d++) acc = fmaf(row[d], w[d], acc);
  q[bu * DD + j] = acc;
}

// ---------------- q2 = q @ W_k  (block per (b,u), thread d) ----------------
__global__ void k_gemm_q2(const float* __restrict__ q, const float* __restrict__ Wk,
                          float* __restrict__ q2) {
  __shared__ float row[DD];
  int bu = blockIdx.x;
  int d = threadIdx.x;
  row[d] = q[bu * DD + d];
  __syncthreads();
  float acc = 0.0f;
  for (int j = 0; j < DD; j++) acc = fmaf(row[j], Wk[j * DD + d], acc);
  q2[bu * DD + d] = acc;
}

// ---------------- pack q2 into WMMA A-fragment layout, f16 hi/lo split -------------
// record index = ((b*64 + ut)*8 + ks)*32 + lane, each record = 16 halves (32B)
// A 16x32 f16 layout (ISA 7.12.2): lane<16: M=lane, K in {0..7, 16..23};
//                                  lane>=16: M=lane-16, K in {8..15, 24..31}
__global__ void k_pack(const float* __restrict__ q2, _Float16* __restrict__ hi,
                       _Float16* __restrict__ lo) {
  int gid = blockIdx.x * 256 + threadIdx.x;   // 65536 records total
  int lane = gid & 31;
  int ks   = (gid >> 5) & 7;
  int ut   = (gid >> 8) & 63;
  int b    = gid >> 14;
  int M = lane & 15;
  int kbase = (lane < 16) ? 0 : 8;
  v16h h, l;
  #pragma unroll
  for (int e = 0; e < 16; e++) {
    int K = (e < 8) ? (kbase + e) : (16 + kbase + (e - 8));
    float v = q2[((b * UU + ut * 16 + M) * DD) + ks * 32 + K];
    _Float16 vh = (_Float16)v;
    h[e] = vh;
    l[e] = (_Float16)(v - (float)vh);
  }
  ((v16h*)hi)[gid] = h;
  ((v16h*)lo)[gid] = l;
}

// ---------------- per-step kernel 1: logits GEMM + server-side reduction -----------
__global__ __launch_bounds__(256) void k_step1(
    const float* __restrict__ server_enc, const unsigned char* __restrict__ connect,
    const float* __restrict__ W_state, const float* __restrict__ b_state,
    const float* __restrict__ W_us,
    const float* __restrict__ p_alpha, const float* __restrict__ p_beta,
    const float* __restrict__ p_sbias,
    const _Float16* __restrict__ q2hi, const _Float16* __restrict__ q2lo,
    const float* __restrict__ needs, const float* __restrict__ cap,
    const float* __restrict__ inv_init, const int* __restrict__ fd_id,
    const unsigned char* __restrict__ allocB, const unsigned* __restrict__ fd_seen,
    const int* __restrict__ rc,
    float* __restrict__ mcol, float* __restrict__ lpbs,
    int* __restrict__ propu, int* __restrict__ validv) {
  int b  = blockIdx.x >> 4;   // 16 server tiles per batch
  int st = blockIdx.x & 15;
  int tid = threadIdx.x;

  __shared__ __align__(32) _Float16 Bhi[8][32][16];  // [ks][lane][e]
  __shared__ __align__(32) _Float16 Blo[8][32][16];
  __shared__ float capS[16][4];
  __shared__ float cratS[16][4];
  __shared__ int   fdS[16];
  __shared__ float mW[8][16];
  __shared__ float sW[8][16];
  __shared__ int   iW[8][16];

  if (tid < 64) {
    int sl = tid >> 2, c = tid & 3;
    int sg = b * SS + st * 16 + sl;
    float cv = cap[sg * 4 + c];
    capS[sl][c]  = cv;
    cratS[sl][c] = cv * inv_init[sg * 4 + c];
  }
  if (tid < 16) fdS[tid] = fd_id[b * SS + st * 16 + tid];
  __syncthreads();

  // Build X_t for 16 servers and pack into LDS as f16 hi/lo B-fragments.
  // B 32x16 f16 layout: lane<16: N=lane, K in {0..7, 16..23}; lane>=16: N=lane-16, K in {8..15, 24..31}
  {
    int d = tid;                 // 0..255
    int ks = d >> 5, kk = d & 31;
    int laneoff = (kk & 8) ? 16 : 0;
    int e = (kk & 7) | ((kk >> 1) & 8);
    float bsv = b_state[d];
    float w0 = W_state[d * 4 + 0], w1 = W_state[d * 4 + 1];
    float w2 = W_state[d * 4 + 2], w3 = W_state[d * 4 + 3];
    for (int sl = 0; sl < 16; sl++) {
      int sg = b * SS + st * 16 + sl;
      float x = server_enc[sg * DD + d] + bsv
              + cratS[sl][0] * w0 + cratS[sl][1] * w1
              + cratS[sl][2] * w2 + cratS[sl][3] * w3;
      _Float16 xh = (_Float16)x;
      Bhi[ks][sl + laneoff][e] = xh;
      Blo[ks][sl + laneoff][e] = (_Float16)(x - (float)xh);
    }
  }
  __syncthreads();

  int wave = tid >> 5, lane = tid & 31;
  int col = lane & 15;
  int sglob = st * 16 + col;
  float c0 = capS[col][0], c1 = capS[col][1], c2 = capS[col][2], c3 = capS[col][3];
  int fdid = fdS[col];
  float alpha = p_alpha[0], beta = p_beta[0], sbias = p_sbias[0];
  float t_new = tanhf(alpha + beta), t_old = tanhf(beta);
  float wu0 = W_us[0], wu1 = W_us[1], wu2 = W_us[2];
  const float inv_sqrt_d = 0.0625f;   // D^-0.5, D=256

  float m = -INFINITY, ssum = 0.0f;
  int idx = 0;

  const v16h* AH = (const v16h*)q2hi;
  const v16h* AL = (const v16h*)q2lo;

  for (int i = 0; i < 8; i++) {       // 8 user-tiles per wave (128 users)
    int ut = wave * 8 + i;
    v8f acc = {0.f, 0.f, 0.f, 0.f, 0.f, 0.f, 0.f, 0.f};
    #pragma unroll
    for (int ks = 0; ks < 8; ks++) {
      int rec = (((b * 64 + ut) * 8) + ks) * 32 + lane;
      v16h ah = AH[rec];
      v16h al = AL[rec];
      v16h bh = *(const v16h*)&Bhi[ks][lane][0];
      v16h bl = *(const v16h*)&Blo[ks][lane][0];
      // f32-accuracy emulation: A = ah+al, B = bh+bl (drop al*bl term)
      acc = __builtin_amdgcn_wmma_f32_16x16x32_f16(false, ah, false, bh, (short)0, acc, false, false);
      acc = __builtin_amdgcn_wmma_f32_16x16x32_f16(false, ah, false, bl, (short)0, acc, false, false);
      acc = __builtin_amdgcn_wmma_f32_16x16x32_f16(false, al, false, bh, (short)0, acc, false, false);
    }
    // fused epilogue + online softmax/argmax for this tile
    int ubase = ut * 16 + ((lane >= 16) ? 8 : 0);
    #pragma unroll
    for (int r = 0; r < 8; r++) {
      int u = ubase + r;
      int bu = b * UU + u;
      float n0 = needs[bu * 4 + 0], n1 = needs[bu * 4 + 1];
      float n2 = needs[bu * 4 + 2], n3 = needs[bu * 4 + 3];
      int rcv = rc[bu];
      unsigned fs = fd_seen[bu];
      unsigned char cn  = connect[(size_t)bu * SS + sglob];
      unsigned char al8 = allocB [(size_t)bu * SS + sglob];
      bool elig = cn && !al8 && (rcv < KREDC) &&
                  (c0 >= n0) && (c1 >= n1) && (c2 >= n2) && (c3 >= n3);
      float rcf = (float)rcv;
      float ubias = wu0 * (rcf * (1.0f / 3.0f))
                  + wu1 * (fmaxf(3.0f - rcf, 0.0f) * (1.0f / 3.0f))
                  + wu2 * ((rcv == KREDC - 1) ? 1.0f : 0.0f);
      float lg = acc[r] * inv_sqrt_d + sbias +
                 (((fs >> fdid) & 1u) ? t_old : t_new) + ubias;
      float cand = elig ? lg : NEGV;
      if (cand > m) { ssum = ssum * __expf(m - cand) + 1.0f; m = cand; idx = u; }
      else          { ssum += __expf(cand - m); }
    }
  }

  // merge lane pair (lane, lane^16) which share a column
  {
    float m2 = __shfl_xor(m, 16, 32);
    float s2 = __shfl_xor(ssum, 16, 32);
    int   i2 = __shfl_xor(idx, 16, 32);
    if (m2 > m || (m2 == m && i2 < idx)) {
      ssum = s2 + ssum * __expf(m - m2);
      m = m2; idx = i2;
    } else {
      ssum = ssum + s2 * __expf(m2 - m);
    }
  }
  if (lane < 16) { mW[wave][col] = m; sW[wave][col] = ssum; iW[wave][col] = idx; }
  __syncthreads();

  if (tid < 16) {
    float M0 = mW[0][tid], S0 = sW[0][tid];
    int I0 = iW[0][tid];
    for (int w = 1; w < 8; w++) {
      float mw = mW[w][tid], sw = sW[w][tid];
      int iw = iW[w][tid];
      if (mw > M0 || (mw == M0 && iw < I0)) { S0 = sw + S0 * __expf(M0 - mw); M0 = mw; I0 = iw; }
      else                                  { S0 += sw * __expf(mw - M0); }
    }
    int sg = b * SS + st * 16 + tid;
    mcol[sg]   = M0;
    lpbs[sg]   = -__logf(S0);
    propu[sg]  = I0;
    validv[sg] = (M0 > -1e8f) ? 1 : 0;
  }
}

// ---------------- per-step kernel 2: user accept + state update ----------------
__global__ void k_step2(const float* __restrict__ mcol, const float* __restrict__ lpbs,
                        const int* __restrict__ propu, const int* __restrict__ validv,
                        const int* __restrict__ fd_id, const float* __restrict__ needs,
                        unsigned char* __restrict__ allocB, unsigned* __restrict__ fd_seen,
                        int* __restrict__ rc, float* __restrict__ cap,
                        float* __restrict__ logp_u) {
  int gid = blockIdx.x * blockDim.x + threadIdx.x;
  if (gid >= BB * UU) return;
  int b = gid >> 10;
  int u = gid & 1023;
  float best = -INFINITY;
  int bs = -1;
  float lacc = 0.0f;
  unsigned bits = 0;
  for (int s = 0; s < SS; s++) {
    int sg = b * SS + s;
    if (validv[sg] && propu[sg] == u) {
      lacc += lpbs[sg];
      bits |= (1u << fd_id[sg]);
      float mv = mcol[sg];
      if (mv > best) { best = mv; bs = s; }   // ascending s + strict > == first max
    }
  }
  if (bs >= 0) {
    allocB[(size_t)gid * SS + bs] = 1;
    rc[gid] += 1;
    int cg = (b * SS + bs) * 4;
    #pragma unroll
    for (int c = 0; c < 4; c++)
      cap[cg + c] = fmaxf(cap[cg + c] - needs[gid * 4 + c], 0.0f);
  }
  fd_seen[gid] |= bits;
  logp_u[gid] += lacc;
}

// ---------------- finalize ----------------
__global__ void k_logp(const float* __restrict__ logp_u, float* __restrict__ out) {
  __shared__ float red[256];
  int b = blockIdx.x, tid = threadIdx.x;
  float a = 0.0f;
  #pragma unroll
  for (int k = 0; k < 4; k++) a += logp_u[b * UU + tid * 4 + k];
  red[tid] = a;
  __syncthreads();
  for (int off = 128; off > 0; off >>= 1) {
    if (tid < off) red[tid] += red[tid + off];
    __syncthreads();
  }
  if (tid == 0) out[b] = red[0];
}

__global__ void k_final(const unsigned char* __restrict__ allocB,
                        const float* __restrict__ cap, float* __restrict__ out) {
  int gid = blockIdx.x * blockDim.x + threadIdx.x;
  if (gid < BB * UU * SS) out[BB + gid] = allocB[gid] ? 1.0f : 0.0f;
  if (gid < BB * SS * 4)  out[BB + BB * UU * SS + gid] = cap[gid];
}

// ---------------- launch ----------------
extern "C" void kernel_launch(void* const* d_in, const int* in_sizes, int n_in,
                              void* d_out, int out_size, void* d_ws, size_t ws_size,
                              hipStream_t stream) {
  (void)in_sizes; (void)n_in; (void)out_size; (void)ws_size;
  const float* user_enc   = (const float*)d_in[0];
  const float* server_enc = (const float*)d_in[1];
  const float* users      = (const float*)d_in[2];
  const float* servers    = (const float*)d_in[3];
  const unsigned char* connect = (const unsigned char*)d_in[4];  // numpy bool = 1 byte
  const float* fd_onehot  = (const float*)d_in[5];
  const float* W_state    = (const float*)d_in[6];
  const float* b_state    = (const float*)d_in[7];
  const float* W_q        = (const float*)d_in[8];
  const float* W_k        = (const float*)d_in[9];
  const float* W_us       = (const float*)d_in[10];
  const float* fd_alpha   = (const float*)d_in[11];
  const float* fd_beta    = (const float*)d_in[12];
  const float* score_bias = (const float*)d_in[13];
  float* out = (float*)d_out;

  char* p = (char*)d_ws;
  auto take = [&](size_t bytes) -> char* {
    char* r = p;
    p += (bytes + 255) & ~((size_t)255);
    return r;
  };
  unsigned char* w_alloc = (unsigned char*)take((size_t)BB * UU * SS);
  unsigned* w_fd   = (unsigned*)take((size_t)BB * UU * 4);
  int*      w_rc   = (int*)take((size_t)BB * UU * 4);
  float*    w_logp = (float*)take((size_t)BB * UU * 4);
  float*    w_cap  = (float*)take((size_t)BB * SS * 4 * 4);
  float*    w_inv  = (float*)take((size_t)BB * SS * 4 * 4);
  int*      w_fdid = (int*)take((size_t)BB * SS * 4);
  float*    w_needs = (float*)take((size_t)BB * UU * 4 * 4);
  float*    w_q    = (float*)take((size_t)BB * UU * DD * 4);
  float*    w_q2   = (float*)take((size_t)BB * UU * DD * 4);
  _Float16* w_hi   = (_Float16*)take((size_t)BB * UU * DD * 2);
  _Float16* w_lo   = (_Float16*)take((size_t)BB * UU * DD * 2);
  float*    w_mcol = (float*)take((size_t)BB * SS * 4);
  float*    w_lpbs = (float*)take((size_t)BB * SS * 4);
  int*      w_prop = (int*)take((size_t)BB * SS * 4);
  int*      w_val  = (int*)take((size_t)BB * SS * 4);

  hipMemsetAsync(w_alloc, 0, (size_t)BB * UU * SS, stream);
  hipMemsetAsync(w_fd,   0, (size_t)BB * UU * 4, stream);
  hipMemsetAsync(w_rc,   0, (size_t)BB * UU * 4, stream);
  hipMemsetAsync(w_logp, 0, (size_t)BB * UU * 4, stream);

  k_init<<<(BB * UU + 255) / 256, 256, 0, stream>>>(users, servers, fd_onehot,
                                                    w_needs, w_cap, w_inv, w_fdid);
  k_gemm_q<<<BB * UU, DD, 0, stream>>>(user_enc, W_q, w_q);
  k_gemm_q2<<<BB * UU, DD, 0, stream>>>(w_q, W_k, w_q2);
  k_pack<<<256, 256, 0, stream>>>(w_q2, w_hi, w_lo);

  for (int step = 0; step < MAXSTEPS; step++) {
    k_step1<<<BB * (SS / 16), 256, 0, stream>>>(
        server_enc, connect, W_state, b_state, W_us,
        fd_alpha, fd_beta, score_bias,
        w_hi, w_lo, w_needs, w_cap, w_inv, w_fdid,
        w_alloc, w_fd, w_rc,
        w_mcol, w_lpbs, w_prop, w_val);
    k_step2<<<(BB * UU + 255) / 256, 256, 0, stream>>>(
        w_mcol, w_lpbs, w_prop, w_val, w_fdid, w_needs,
        w_alloc, w_fd, w_rc, w_cap, w_logp);
  }

  k_logp<<<BB, 256, 0, stream>>>(w_logp, out);
  k_final<<<(BB * UU * SS + 255) / 256, 256, 0, stream>>>(w_alloc, w_cap, out);
}
